// BlockG_23467701305349
// MI455X (gfx1250) — matrix operand order repaired and checked by reading, not compiled, not run on hardware
//
#include <hip/hip_runtime.h>
#include <hip/hip_bf16.h>

// ---------------------------------------------------------------------------
// StyleGAN2-like block on gfx1250 (MI455X).
// conv(x, W*s)*d == conv(x*s, W)*d  -> batch-invariant weights, implicit GEMM
// over 9 taps with v_wmma_f32_16x16x32_bf16, fragment-major LDS staging.
// ---------------------------------------------------------------------------

typedef unsigned short u16;
typedef unsigned int   u32;
typedef __attribute__((ext_vector_type(16))) __bf16 v16bf;
typedef __attribute__((ext_vector_type(16))) u16    v16u;
typedef __attribute__((ext_vector_type(8)))  float  v8f;

#define B_  8
#define C_  512
#define HW_ 4096   // 64*64
#define W_  64
#define EPS 1e-8f
#define SLOPE 0.2f

__device__ __forceinline__ u16 f2bf(float f) {
  unsigned int u = __builtin_bit_cast(unsigned int, f);
  u += 0x7FFFu + ((u >> 16) & 1u);            // round-to-nearest-even
  return (u16)(u >> 16);
}
__device__ __forceinline__ float bf2f(u16 h) {
  return __builtin_bit_cast(float, ((unsigned int)h) << 16);
}
__device__ __forceinline__ float leaky(float v) {
  return v >= 0.f ? v : SLOPE * v;
}
// bilinear x2 upsample, align_corners=True, n=32 source
__device__ __forceinline__ float up2s(const float* src, int oy, int ox) {
  const float sc = 31.0f / 63.0f;
  float cy = oy * sc, cx = ox * sc;
  int iy0 = (int)cy, ix0 = (int)cx;
  int iy1 = iy0 + 1 < 32 ? iy0 + 1 : 31;
  int ix1 = ix0 + 1 < 32 ? ix0 + 1 : 31;
  float ty = cy - (float)iy0, tx = cx - (float)ix0;
  float v00 = src[iy0 * 32 + ix0], v01 = src[iy0 * 32 + ix1];
  float v10 = src[iy1 * 32 + ix0], v11 = src[iy1 * 32 + ix1];
  return (1.f - ty) * ((1.f - tx) * v00 + tx * v01) +
         ty * ((1.f - tx) * v10 + tx * v11);
}

// ---- 1) styles_l = w @ A_l^T + b_l, for l=1..3 ------------------------------
__global__ void k_styles(const float* __restrict__ w,
                         const float* __restrict__ A1w, const float* __restrict__ A1b,
                         const float* __restrict__ A2w, const float* __restrict__ A2b,
                         const float* __restrict__ A3w, const float* __restrict__ A3b,
                         float* __restrict__ s1, float* __restrict__ s2,
                         float* __restrict__ s3) {
  int idx = blockIdx.x * blockDim.x + threadIdx.x;
  if (idx >= 3 * B_ * C_) return;
  int l = idx >> 12;              // /4096
  int b = (idx >> 9) & 7;
  int o = idx & 511;
  const float* Aw = l == 0 ? A1w : (l == 1 ? A2w : A3w);
  const float* Ab = l == 0 ? A1b : (l == 1 ? A2b : A3b);
  float* s = l == 0 ? s1 : (l == 1 ? s2 : s3);
  float acc = Ab[o];
  const float* wr = w + b * C_;
  const float* ar = Aw + o * C_;
  for (int i = 0; i < C_; ++i) acc += wr[i] * ar[i];
  s[b * C_ + o] = acc;
}

// ---- 2) S_l[co,ci] = sum_k W_l[co,ci,k]^2 ----------------------------------
__global__ void k_wsq(const float* __restrict__ W1, const float* __restrict__ W2,
                      float* __restrict__ S1, float* __restrict__ S2) {
  int idx = blockIdx.x * blockDim.x + threadIdx.x;
  if (idx >= 2 * C_ * C_) return;
  int l = idx / (C_ * C_);
  int j = idx - l * (C_ * C_);
  const float* W = l ? W2 : W1;
  const float* p = W + (size_t)j * 9;
  float acc = 0.f;
#pragma unroll
  for (int k = 0; k < 9; ++k) acc += p[k] * p[k];
  (l ? S2 : S1)[j] = acc;
}

// ---- 3) demod factors -------------------------------------------------------
__global__ void k_demod(const float* __restrict__ S1, const float* __restrict__ S2,
                        const float* __restrict__ s1, const float* __restrict__ s2,
                        const float* __restrict__ s3, const float* __restrict__ W3,
                        float* __restrict__ d1, float* __restrict__ d2,
                        float* __restrict__ d3) {
  int idx = blockIdx.x * blockDim.x + threadIdx.x;
  if (idx < 2 * B_ * C_) {
    int l = idx >> 12;
    int b = (idx >> 9) & 7;
    int co = idx & 511;
    const float* S = l ? S2 : S1;
    const float* s = l ? s2 : s1;
    const float* Sr = S + co * C_;
    const float* sr = s + b * C_;
    float acc = 0.f;
    for (int ci = 0; ci < C_; ++ci) { float sv = sr[ci]; acc += Sr[ci] * sv * sv; }
    (l ? d2 : d1)[b * C_ + co] = rsqrtf(acc + EPS);
  } else if (idx < 2 * B_ * C_ + B_ * 3) {
    int j = idx - 2 * B_ * C_;
    int b = j / 3, c = j - 3 * b;
    const float* wr = W3 + c * C_;
    const float* sr = s3 + b * C_;
    float acc = 0.f;
    for (int ci = 0; ci < C_; ++ci) { float t = wr[ci] * sr[ci]; acc += t * t; }
    d3[b * 3 + c] = rsqrtf(acc + EPS);
  }
}

// ---- 4) repack weights to bf16, tap-major: Wb[tap][co][ci] ------------------
__global__ void k_wconvert(const float* __restrict__ W1, const float* __restrict__ W2,
                           u16* __restrict__ Wb1, u16* __restrict__ Wb2) {
  int idx = blockIdx.x * blockDim.x + threadIdx.x;
  const int per = 9 * C_ * C_;
  if (idx >= 2 * per) return;
  int l = idx / per;
  int j = idx - l * per;
  int tap = j / (C_ * C_);
  int rem = j - tap * (C_ * C_);
  int co = rem >> 9, ci = rem & 511;
  const float* W = l ? W2 : W1;
  float v = W[((size_t)(co * C_ + ci)) * 9 + tap];
  (l ? Wb2 : Wb1)[((size_t)(tap * C_ + co)) * C_ + ci] = f2bf(v);
}

// ---- 5) up2(x) * style1 -> bf16 act1 ---------------------------------------
__global__ void k_upmod(const float* __restrict__ x, const float* __restrict__ s1,
                        u16* __restrict__ act1) {
  int idx = blockIdx.x * blockDim.x + threadIdx.x;
  if (idx >= B_ * C_ * HW_) return;
  int p = idx & (HW_ - 1);
  int c = (idx >> 12) & 511;
  int b = idx >> 21;
  int oy = p >> 6, ox = p & 63;
  float v = up2s(x + ((size_t)(b * C_ + c)) * 1024, oy, ox);
  act1[idx] = f2bf(v * s1[b * C_ + c]);
}

// ---- 6/7) implicit-GEMM modulated conv 3x3 (WMMA) ---------------------------
// grid = (C/128, 64 rows, B); block = 256 (8 wave32s, 4x2 wave grid)
// LDS holds A/B tiles already in WMMA fragment order: one lane's 16 bf16
// elements are 32 contiguous bytes -> fragment read = single 32B LDS read.
__global__ __launch_bounds__(256)
void k_conv(const u16* __restrict__ Wb, const u16* __restrict__ act_in,
            const float* __restrict__ dmod, const float* __restrict__ bias,
            const float* __restrict__ noise, const float* __restrict__ nstyle,
            u16* __restrict__ act_out, float* __restrict__ out_f32,
            int write_f32) {
  // AF[g][lane][0..1]: A subtile g (16 co rows x 32 K), 32B fragment per lane
  // BF[g][lane][0..1]: B subtile g (32 K x 16 px),      32B fragment per lane
  __shared__ __align__(32) uint4 AF[8][32][2];   // 8 KB
  __shared__ __align__(32) uint4 BF[4][32][2];   // 4 KB

  const int tid = threadIdx.x;
  const int co0 = blockIdx.x * 128;
  const int y   = blockIdx.y;
  const int b   = blockIdx.z;
  const int lane = tid & 31;
  const int waveId = tid >> 5;
  const int wm = waveId & 3;     // 4 waves along M
  const int wn = waveId >> 2;    // 2 waves along N
  const int mrow = lane & 15;
  const int halfl = lane >> 4;

  // A staging ids: thread -> (row r of 128, K half hseg)
  const int ar = tid >> 1;
  const int ahseg = tid & 1;
  // B staging ids: thread -> (pixel x, row-group rgrp of 8 ci rows)
  const int bx = tid & 63;
  const int brgrp = tid >> 6;                 // 0..3
  const int blane = (bx & 15) + 16 * (brgrp & 1);
  const int bg = bx >> 4;                     // which 16-col subtile
  const int bhalf = brgrp >> 1;               // which 16B of the fragment

  v8f acc[2][2] = {};

  for (int tap = 0; tap < 9; ++tap) {
    const int dy = tap / 3 - 1;
    const int dx = tap % 3 - 1;
    const int ys = y + dy;
    const unsigned yok = (unsigned)ys < 64u ? 1u : 0u;
    const int ysc = ys < 0 ? 0 : (ys > 63 ? 63 : ys);
    const int xs = bx + dx;
    const unsigned xok = (unsigned)xs < 64u ? 1u : 0u;
    const int xsc = xs < 0 ? 0 : (xs > 63 ? 63 : xs);
    const u16 msk = (u16)(0u - (yok & xok));
    const int boff = ysc * W_ + xsc;

    for (int k0 = 0; k0 < C_; k0 += 32) {
      // --- stage A: 2x b128 global load -> 2x b128 LDS store (pre-swizzled)
      {
        const u16* src = Wb + (((size_t)(tap * C_ + co0 + ar)) * C_) + k0 + ahseg * 16;
        if (k0 + 32 < C_) __builtin_prefetch(src + 32, 0, 3);  // global_prefetch_b8
        uint4 lo = *(const uint4*)src;        // K = ks+0..7
        uint4 hi = *(const uint4*)(src + 8);  // K = ks+8..15
        AF[ar >> 4][(ar & 15) + 0][ahseg]  = lo;   // half 0 of lane group
        AF[ar >> 4][(ar & 15) + 16][ahseg] = hi;   // half 1 of lane group
      }
      // --- stage B: 8 masked u16 loads (fixed x, 8 ci rows) -> 1x b128 store
      {
        const u16* abase = act_in + ((size_t)(b * C_ + k0 + brgrp * 8)) * HW_ + boff;
        u16 v[8];
#pragma unroll
        for (int i = 0; i < 8; ++i)
          v[i] = (u16)(abase[(size_t)i * HW_] & msk);
        uint4 pk;
        pk.x = (u32)v[0] | ((u32)v[1] << 16);
        pk.y = (u32)v[2] | ((u32)v[3] << 16);
        pk.z = (u32)v[4] | ((u32)v[5] << 16);
        pk.w = (u32)v[6] | ((u32)v[7] << 16);
        BF[bg][blane][bhalf] = pk;
      }
      __syncthreads();

      // --- fragments: single 32B LDS read each, then 4 WMMAs ---------------
      v16u au[2], bu[2];
#pragma unroll
      for (int mi = 0; mi < 2; ++mi)
        au[mi] = *(const v16u*)&AF[wm * 2 + mi][lane][0];
#pragma unroll
      for (int ni = 0; ni < 2; ++ni)
        bu[ni] = *(const v16u*)&BF[wn * 2 + ni][lane][0];
#pragma unroll
      for (int mi = 0; mi < 2; ++mi)
#pragma unroll
        for (int ni = 0; ni < 2; ++ni)
          acc[mi][ni] = __builtin_amdgcn_wmma_f32_16x16x32_bf16(
              false, __builtin_bit_cast(v16bf, au[mi]),
              false, __builtin_bit_cast(v16bf, bu[ni]),
              (short)0, acc[mi][ni], false, false);
      __syncthreads();
    }
  }

  // --- epilogue: *demod + bias*noise, leaky, write f32 and/or next bf16 -----
#pragma unroll
  for (int mi = 0; mi < 2; ++mi) {
#pragma unroll
    for (int ni = 0; ni < 2; ++ni) {
      int xcol = wn * 32 + ni * 16 + mrow;
      float nz = noise[b * HW_ + y * W_ + xcol];
#pragma unroll
      for (int r = 0; r < 8; ++r) {
        int co = co0 + wm * 32 + mi * 16 + r + 8 * halfl;
        float val = acc[mi][ni][r] * dmod[b * C_ + co] + bias[co] * nz;
        val = leaky(val);
        size_t oidx = ((size_t)(b * C_ + co)) * HW_ + y * W_ + xcol;
        if (write_f32) out_f32[oidx] = val;
        act_out[oidx] = f2bf(val * nstyle[b * C_ + co]);
      }
    }
  }
}

// ---- 8) toRGB: 1x1 conv(act3, W3)*d3 + B3*noise3, leaky, + up2(rgb) --------
__global__ void k_torgb(const u16* __restrict__ act3, const float* __restrict__ W3,
                        const float* __restrict__ d3, const float* __restrict__ B3,
                        const float* __restrict__ noise3, const float* __restrict__ rgb,
                        float* __restrict__ rgb_out) {
  int idx = blockIdx.x * blockDim.x + threadIdx.x;
  if (idx >= B_ * 3 * HW_) return;
  int p = idx & (HW_ - 1);
  int c = (idx >> 12) % 3;
  int b = idx / (3 * HW_);
  const u16* a = act3 + (size_t)b * C_ * HW_ + p;
  const float* wr = W3 + c * C_;
  float acc = 0.f;
  for (int ci = 0; ci < C_; ++ci) acc += wr[ci] * bf2f(a[(size_t)ci * HW_]);
  float val = leaky(acc * d3[b * 3 + c] + B3[c] * noise3[b * HW_ + p]);
  float base = up2s(rgb + ((size_t)(b * 3 + c)) * 1024, p >> 6, p & 63);
  rgb_out[idx] = base + val;
}

// ---------------------------------------------------------------------------
extern "C" void kernel_launch(void* const* d_in, const int* in_sizes, int n_in,
                              void* d_out, int out_size, void* d_ws, size_t ws_size,
                              hipStream_t stream) {
  const float* x      = (const float*)d_in[0];
  const float* rgb    = (const float*)d_in[1];
  const float* w      = (const float*)d_in[2];
  const float* noise1 = (const float*)d_in[3];
  const float* noise2 = (const float*)d_in[4];
  const float* noise3 = (const float*)d_in[5];
  const float* W1     = (const float*)d_in[6];
  const float* A1w    = (const float*)d_in[7];
  const float* A1b    = (const float*)d_in[8];
  const float* B1     = (const float*)d_in[9];
  const float* W2     = (const float*)d_in[10];
  const float* A2w    = (const float*)d_in[11];
  const float* A2b    = (const float*)d_in[12];
  const float* B2     = (const float*)d_in[13];
  const float* W3     = (const float*)d_in[14];
  const float* A3w    = (const float*)d_in[15];
  const float* A3b    = (const float*)d_in[16];
  const float* B3     = (const float*)d_in[17];

  float* out_x   = (float*)d_out;                       // [8,512,64,64]
  float* out_rgb = out_x + (size_t)B_ * C_ * HW_;       // [8,3,64,64]

  // workspace partition (256B aligned slices)
  size_t cur = 0;
  auto take = [&](size_t bytes) -> void* {
    void* p = (char*)d_ws + cur;
    cur += (bytes + 255) & ~(size_t)255;
    return p;
  };
  float* s1 = (float*)take(B_ * C_ * 4);
  float* s2 = (float*)take(B_ * C_ * 4);
  float* s3 = (float*)take(B_ * C_ * 4);
  float* d1 = (float*)take(B_ * C_ * 4);
  float* d2 = (float*)take(B_ * C_ * 4);
  float* d3 = (float*)take(B_ * 3 * 4);
  float* S1 = (float*)take((size_t)C_ * C_ * 4);
  float* S2 = (float*)take((size_t)C_ * C_ * 4);
  u16* Wb1  = (u16*)take((size_t)9 * C_ * C_ * 2);
  u16* Wb2  = (u16*)take((size_t)9 * C_ * C_ * 2);
  u16* act1 = (u16*)take((size_t)B_ * C_ * HW_ * 2);    // reused as act3
  u16* act2 = (u16*)take((size_t)B_ * C_ * HW_ * 2);
  u16* act3 = act1;                                     // act1 dead after conv1

  const int T = 256;
  k_styles<<<(3 * B_ * C_ + T - 1) / T, T, 0, stream>>>(
      w, A1w, A1b, A2w, A2b, A3w, A3b, s1, s2, s3);
  k_wsq<<<(2 * C_ * C_ + T - 1) / T, T, 0, stream>>>(W1, W2, S1, S2);
  k_demod<<<(2 * B_ * C_ + B_ * 3 + T - 1) / T, T, 0, stream>>>(
      S1, S2, s1, s2, s3, W3, d1, d2, d3);
  k_wconvert<<<(2 * 9 * C_ * C_ + T - 1) / T, T, 0, stream>>>(W1, W2, Wb1, Wb2);
  k_upmod<<<(B_ * C_ * HW_ + T - 1) / T, T, 0, stream>>>(x, s1, act1);

  dim3 cgrid(C_ / 128, 64, B_);
  // layer 1: reads act1, writes act2 = leaky(conv)*s2 (bf16 only)
  k_conv<<<cgrid, T, 0, stream>>>(Wb1, act1, d1, B1, noise1, s2,
                                  act2, out_x, /*write_f32=*/0);
  // layer 2: reads act2, writes x (f32) and act3 = x*s3 (bf16)
  k_conv<<<cgrid, T, 0, stream>>>(Wb2, act2, d2, B2, noise2, s3,
                                  act3, out_x, /*write_f32=*/1);

  k_torgb<<<(B_ * 3 * HW_ + T - 1) / T, T, 0, stream>>>(
      act3, W3, d3, B3, noise3, rgb, out_rgb);
}
